// HierarchicalMoE_40355512714068
// MI455X (gfx1250) — matrix operand order repaired and checked by reading, not compiled, not run on hardware
//
#include <hip/hip_runtime.h>
#include <stdint.h>

#define N_TOK 4096
#define DIM   1024
#define NG    4
#define NE    4
#define NF    4096
#define NEXP  16   // G*E
#define MT    32   // tokens per FFN block (2 x 16-row WMMA subtiles)

typedef __attribute__((ext_vector_type(16))) __bf16 v16bf;
typedef __attribute__((ext_vector_type(2)))  __bf16 v2bf;
typedef __attribute__((ext_vector_type(8)))  float  v8f;

union ABFrag { v16bf v; uint32_t u[8]; };

__device__ __forceinline__ uint32_t packbf2(float lo, float hi) {
  union { v2bf v; uint32_t u; } c;
  c.v[0] = (__bf16)lo;
  c.v[1] = (__bf16)hi;
  return c.u;
}
__device__ __forceinline__ float gelu_exact(float x) {
  return 0.5f * x * (1.0f + erff(x * 0.70710678118654752f));
}

// ---------------- K0: zero entropy + expert counters ----------------
__global__ void init_kernel(float* ent, int* ecount) {
  if (threadIdx.x == 0) *ent = 0.0f;
  if (threadIdx.x < NEXP) ecount[threadIdx.x] = 0;
}

// ---------------- K1: router (one wave per token) ----------------
__global__ void __launch_bounds__(256)
router_kernel(const float* __restrict__ x, const float* __restrict__ Wg,
              const float* __restrict__ We, float* __restrict__ glog,
              float* __restrict__ ent, int* __restrict__ route_gi,
              float* __restrict__ route_w) {
  int wave = threadIdx.x >> 5;
  int lane = threadIdx.x & 31;
  int n = blockIdx.x * 8 + wave;
  if (n >= N_TOK) return;
  const float* xr = x + (size_t)n * DIM;
  float ag[NG] = {0.f, 0.f, 0.f, 0.f};
  float ae[NEXP];
#pragma unroll
  for (int k = 0; k < NEXP; k++) ae[k] = 0.f;
  for (int d = lane; d < DIM; d += 32) {
    float xv = xr[d];
#pragma unroll
    for (int g = 0; g < NG; g++) ag[g] += xv * Wg[g * DIM + d];
#pragma unroll
    for (int k = 0; k < NEXP; k++) ae[k] += xv * We[(size_t)k * DIM + d];
  }
#pragma unroll
  for (int off = 16; off >= 1; off >>= 1) {
#pragma unroll
    for (int g = 0; g < NG; g++) ag[g] += __shfl_xor(ag[g], off, 32);
#pragma unroll
    for (int k = 0; k < NEXP; k++) ae[k] += __shfl_xor(ae[k], off, 32);
  }
  if (lane == 0) {
#pragma unroll
    for (int g = 0; g < NG; g++) glog[n * NG + g] = ag[g];
    float m = fmaxf(fmaxf(ag[0], ag[1]), fmaxf(ag[2], ag[3]));
    float e[NG], s = 0.f;
#pragma unroll
    for (int g = 0; g < NG; g++) { e[g] = __expf(ag[g] - m); s += e[g]; }
    float ls = __logf(s), h = 0.f;
#pragma unroll
    for (int g = 0; g < NG; g++) { float p = e[g] / s; h -= p * ((ag[g] - m) - ls); }
    atomicAdd(ent, h * (1.0f / (float)N_TOK));
    int g0 = 0;
    for (int g = 1; g < NG; g++) if (ag[g] > ag[g0]) g0 = g;
    int g1 = -1;
    for (int g = 0; g < NG; g++) { if (g == g0) continue; if (g1 < 0 || ag[g] > ag[g1]) g1 = g; }
    float w0 = 1.0f / (1.0f + __expf(ag[g1] - ag[g0]));
    float w1 = 1.0f - w0;
    int e0 = 0, e1 = 0;
    for (int k = 1; k < NE; k++) if (ae[g0 * NE + k] > ae[g0 * NE + e0]) e0 = k;
    for (int k = 1; k < NE; k++) if (ae[g1 * NE + k] > ae[g1 * NE + e1]) e1 = k;
    route_gi[n * 4 + 0] = g0; route_gi[n * 4 + 1] = e0;
    route_gi[n * 4 + 2] = g1; route_gi[n * 4 + 3] = e1;
    route_w[n * 2 + 0] = w0;  route_w[n * 2 + 1] = w1;
  }
}

// ---------------- K2: deterministic per-expert token lists ----------------
__global__ void __launch_bounds__(256)
build_lists_kernel(const int* __restrict__ route_gi, const float* __restrict__ route_w,
                   int* __restrict__ ecount, int* __restrict__ tlist,
                   float* __restrict__ wlist, int* __restrict__ rlist) {
  __shared__ int scan[256];
  __shared__ int sbase;
  const int ex = blockIdx.x, g = ex >> 2, e = ex & 3;
  const int tid = threadIdx.x;
  if (tid == 0) sbase = 0;
  __syncthreads();
  for (int chunk = 0; chunk < N_TOK; chunk += 256) {
    int n = chunk + tid;
    int rank = -1; float wt = 0.f;
    if (route_gi[n * 4 + 0] == g && route_gi[n * 4 + 1] == e) { rank = 0; wt = route_w[n * 2 + 0]; }
    else if (route_gi[n * 4 + 2] == g && route_gi[n * 4 + 3] == e) { rank = 1; wt = route_w[n * 2 + 1]; }
    int flag = (rank >= 0) ? 1 : 0;
    scan[tid] = flag;
    __syncthreads();
    for (int off = 1; off < 256; off <<= 1) {
      int v = scan[tid];
      int add = (tid >= off) ? scan[tid - off] : 0;
      __syncthreads();
      scan[tid] = v + add;
      __syncthreads();
    }
    int incl = scan[tid];
    int total = scan[255];
    int base = sbase;
    if (flag) {
      int pos = base + incl - 1;
      tlist[ex * N_TOK + pos] = n;
      wlist[ex * N_TOK + pos] = wt;
      rlist[ex * N_TOK + pos] = rank;
    }
    __syncthreads();
    if (tid == 0) sbase = base + total;
    __syncthreads();
  }
  if (tid == 0) ecount[ex] = sbase;
}

// ---------------- K3: fp32 -> bf16 staging (x and, optionally, weights) ----------------
__global__ void cvt_bf16_kernel(const float* __restrict__ src, uint32_t* __restrict__ dst,
                                long total2) {
  long stride = (long)gridDim.x * blockDim.x;
  for (long j = (long)blockIdx.x * blockDim.x + threadIdx.x; j < total2; j += stride) {
    float2 v = ((const float2*)src)[j];
    dst[j] = packbf2(v.x, v.y);
  }
}

// ---------------- K4: grouped FFN GEMM (bf16 WMMA, fused GELU, M=32) ----------------
// PREW=true: weights pre-converted to bf16 in workspace (pure b128 loads, no cvt VALU)
template <bool PREW>
__global__ void __launch_bounds__(256)
ffn_kernel(const unsigned short* __restrict__ xbf,
           const float* __restrict__ W1, const float* __restrict__ W2,
           const unsigned short* __restrict__ W1b, const unsigned short* __restrict__ W2b,
           const int* __restrict__ ecount, const int* __restrict__ tlist,
           const float* __restrict__ wlist, const int* __restrict__ rlist,
           float* __restrict__ partial) {
  __shared__ __align__(16) __bf16 xs[MT * DIM];   // 64 KB bf16 x-tile (32 tokens)
  __shared__ __align__(16) __bf16 hs[MT * 128];   // 8 KB bf16 h-chunk
  __shared__ int stok[MT]; __shared__ float swt[MT]; __shared__ int srk[MT];

  const int ex = blockIdx.x;
  const int cnt = ecount[ex];
  const int row0 = (int)blockIdx.y * MT;
  if (row0 >= cnt) return;                      // uniform across block
  const int rows = min(MT, cnt - row0);
  const int tid = threadIdx.x;
  const int wv = tid >> 5;
  const int lane = tid & 31;
  const int hi = lane >> 4;                     // 0: lanes 0-15, 1: lanes 16-31
  const int ln = lane & 15;

  if (tid < MT) {
    if (tid < rows) {
      stok[tid] = tlist[ex * N_TOK + row0 + tid];
      swt[tid]  = wlist[ex * N_TOK + row0 + tid];
      srk[tid]  = rlist[ex * N_TOK + row0 + tid];
    } else { stok[tid] = 0; swt[tid] = 0.f; srk[tid] = 0; }
  }
  __syncthreads();
  { // stage x-tile: 4096 uint4 (32 rows x 128), 16 per thread, fully coalesced
    uint4* xsv = (uint4*)xs;
    for (int i = tid; i < MT * 128; i += 256) {
      int r = i >> 7, c = i & 127;
      uint4 v = {0u, 0u, 0u, 0u};
      if (r < rows) v = ((const uint4*)(xbf + (size_t)stok[r] * DIM))[c];
      xsv[i] = v;
    }
  }
  __syncthreads();

  const uint32_t* xs32 = (const uint32_t*)xs;
  const uint32_t* hs32 = (const uint32_t*)hs;

  v8f zero = {};
  v8f acc2[2][8];                               // 32 x 128 persistent out tile per wave
#pragma unroll
  for (int s = 0; s < 2; s++)
#pragma unroll
    for (int t = 0; t < 8; t++) acc2[s][t] = zero;

  for (int fc = 0; fc < NF; fc += 128) {
    // ---- Phase A: h-tiles (2 x 16) x 16 per wave, K over D
    v8f accA[2] = {zero, zero};
    const int fcol0 = fc + wv * 16;
    for (int d0 = 0; d0 < DIM; d0 += 32) {
      ABFrag a0, a1, b;
      {
        const uint4* ap = (const uint4*)(xs32 + ln * 512 + (d0 >> 1) + hi * 8);
        uint4 q0 = ap[0], q1 = ap[1];
        a0.u[0] = q0.x; a0.u[1] = q0.y; a0.u[2] = q0.z; a0.u[3] = q0.w;
        a0.u[4] = q1.x; a0.u[5] = q1.y; a0.u[6] = q1.z; a0.u[7] = q1.w;
        const uint4* ap2 = (const uint4*)(xs32 + (ln + 16) * 512 + (d0 >> 1) + hi * 8);
        uint4 q2 = ap2[0], q3 = ap2[1];
        a1.u[0] = q2.x; a1.u[1] = q2.y; a1.u[2] = q2.z; a1.u[3] = q2.w;
        a1.u[4] = q3.x; a1.u[5] = q3.y; a1.u[6] = q3.z; a1.u[7] = q3.w;
      }
      if (PREW) {
        const uint4* wp = (const uint4*)(W1b + (size_t)ex * NF * DIM
                                         + (size_t)(fcol0 + ln) * DIM + d0 + hi * 16);
        uint4 p0 = wp[0], p1 = wp[1];
        b.u[0] = p0.x; b.u[1] = p0.y; b.u[2] = p0.z; b.u[3] = p0.w;
        b.u[4] = p1.x; b.u[5] = p1.y; b.u[6] = p1.z; b.u[7] = p1.w;
      } else {
        const float4* wp = (const float4*)(W1 + (size_t)ex * NF * DIM
                                           + (size_t)(fcol0 + ln) * DIM + d0 + hi * 16);
#pragma unroll
        for (int q = 0; q < 4; q++) {
          float4 f4 = wp[q];
          b.u[q * 2 + 0] = packbf2(f4.x, f4.y);
          b.u[q * 2 + 1] = packbf2(f4.z, f4.w);
        }
      }
      accA[0] = __builtin_amdgcn_wmma_f32_16x16x32_bf16(false, a0.v, false, b.v,
                                                        (short)0, accA[0], false, false);
      accA[1] = __builtin_amdgcn_wmma_f32_16x16x32_bf16(false, a1.v, false, b.v,
                                                        (short)0, accA[1], false, false);
    }
#pragma unroll
    for (int s = 0; s < 2; s++)
#pragma unroll
      for (int r = 0; r < 8; r++) {             // C layout: M=r+hi*8, N=ln
        int M = s * 16 + r + hi * 8;
        hs[M * 128 + wv * 16 + ln] = (__bf16)gelu_exact(accA[s][r]);
      }
    __syncthreads();
    // ---- Phase B: out[32 x 128 per wave] += h_chunk @ W2_chunk^T
    for (int k0 = 0; k0 < 128; k0 += 32) {
      ABFrag a0, a1;
      {
        const uint4* ap = (const uint4*)(hs32 + ln * 64 + (k0 >> 1) + hi * 8);
        uint4 q0 = ap[0], q1 = ap[1];
        a0.u[0] = q0.x; a0.u[1] = q0.y; a0.u[2] = q0.z; a0.u[3] = q0.w;
        a0.u[4] = q1.x; a0.u[5] = q1.y; a0.u[6] = q1.z; a0.u[7] = q1.w;
        const uint4* ap2 = (const uint4*)(hs32 + (ln + 16) * 64 + (k0 >> 1) + hi * 8);
        uint4 q2 = ap2[0], q3 = ap2[1];
        a1.u[0] = q2.x; a1.u[1] = q2.y; a1.u[2] = q2.z; a1.u[3] = q2.w;
        a1.u[4] = q3.x; a1.u[5] = q3.y; a1.u[6] = q3.z; a1.u[7] = q3.w;
      }
#pragma unroll
      for (int t = 0; t < 8; t++) {
        ABFrag b;
        if (PREW) {
          const uint4* wp = (const uint4*)(W2b + (size_t)ex * DIM * NF
                                           + (size_t)(wv * 128 + t * 16 + ln) * NF
                                           + fc + k0 + hi * 16);
          uint4 p0 = wp[0], p1 = wp[1];
          b.u[0] = p0.x; b.u[1] = p0.y; b.u[2] = p0.z; b.u[3] = p0.w;
          b.u[4] = p1.x; b.u[5] = p1.y; b.u[6] = p1.z; b.u[7] = p1.w;
        } else {
          const float4* wp = (const float4*)(W2 + (size_t)ex * DIM * NF
                                             + (size_t)(wv * 128 + t * 16 + ln) * NF
                                             + fc + k0 + hi * 16);
#pragma unroll
          for (int q = 0; q < 4; q++) {
            float4 f4 = wp[q];
            b.u[q * 2 + 0] = packbf2(f4.x, f4.y);
            b.u[q * 2 + 1] = packbf2(f4.z, f4.w);
          }
        }
        acc2[0][t] = __builtin_amdgcn_wmma_f32_16x16x32_bf16(false, a0.v, false, b.v,
                                                             (short)0, acc2[0][t], false, false);
        acc2[1][t] = __builtin_amdgcn_wmma_f32_16x16x32_bf16(false, a1.v, false, b.v,
                                                             (short)0, acc2[1][t], false, false);
      }
    }
    __syncthreads();
  }
  // Scatter weighted result to per-(token, rank) partial slot (deterministic)
#pragma unroll
  for (int s = 0; s < 2; s++)
#pragma unroll
    for (int t = 0; t < 8; t++) {
      int col = wv * 128 + t * 16 + ln;
#pragma unroll
      for (int r = 0; r < 8; r++) {
        int M = s * 16 + r + hi * 8;
        if (M < rows) {
          size_t o = ((size_t)stok[M] * 2 + srk[M]) * DIM + col;
          partial[o] = swt[M] * acc2[s][t][r];
        }
      }
    }
}

// ---------------- K5: combine the two expert contributions ----------------
__global__ void combine_kernel(const float* __restrict__ partial, float* __restrict__ out) {
  const int total = N_TOK * DIM;
  for (int j = blockIdx.x * blockDim.x + threadIdx.x; j < total; j += gridDim.x * blockDim.x) {
    int n = j >> 10, d = j & (DIM - 1);
    size_t b = (size_t)n * 2 * DIM + d;
    out[j] = partial[b] + partial[b + DIM];
  }
}

extern "C" void kernel_launch(void* const* d_in, const int* in_sizes, int n_in,
                              void* d_out, int out_size, void* d_ws, size_t ws_size,
                              hipStream_t stream) {
  (void)in_sizes; (void)n_in; (void)out_size;
  const float* x  = (const float*)d_in[0];
  const float* Wg = (const float*)d_in[1];
  const float* We = (const float*)d_in[2];
  const float* W1 = (const float*)d_in[3];
  const float* W2 = (const float*)d_in[4];
  float* out  = (float*)d_out;
  float* glog = out + (size_t)N_TOK * DIM;
  float* ent  = glog + (size_t)N_TOK * NG;

  char* w = (char*)d_ws;
  auto alloc = [&](size_t bytes) -> char* {
    char* p = w; w += (bytes + 255) & ~(size_t)255; return p;
  };
  int*            route_gi = (int*)           alloc((size_t)N_TOK * 4 * sizeof(int));
  float*          route_w  = (float*)         alloc((size_t)N_TOK * 2 * sizeof(float));
  int*            ecount   = (int*)           alloc(NEXP * sizeof(int));
  int*            tlist    = (int*)           alloc((size_t)NEXP * N_TOK * sizeof(int));
  float*          wlist    = (float*)         alloc((size_t)NEXP * N_TOK * sizeof(float));
  int*            rlist    = (int*)           alloc((size_t)NEXP * N_TOK * sizeof(int));
  unsigned short* xbf      = (unsigned short*)alloc((size_t)N_TOK * DIM * sizeof(unsigned short));
  float*          partial  = (float*)         alloc((size_t)N_TOK * 2 * DIM * sizeof(float));

  const size_t welems = (size_t)NEXP * NF * DIM;            // 67.1M per matrix
  size_t base_used = (size_t)(w - (char*)d_ws);
  bool prew = (ws_size >= base_used + 2 * welems * sizeof(unsigned short) + 512);
  unsigned short* W1b = nullptr;
  unsigned short* W2b = nullptr;
  if (prew) {
    W1b = (unsigned short*)alloc(welems * sizeof(unsigned short));
    W2b = (unsigned short*)alloc(welems * sizeof(unsigned short));
  }

  hipLaunchKernelGGL(init_kernel,        dim3(1),         dim3(32),  0, stream, ent, ecount);
  hipLaunchKernelGGL(router_kernel,      dim3(N_TOK / 8), dim3(256), 0, stream,
                     x, Wg, We, glog, ent, route_gi, route_w);
  hipLaunchKernelGGL(build_lists_kernel, dim3(NEXP),      dim3(256), 0, stream,
                     route_gi, route_w, ecount, tlist, wlist, rlist);
  hipLaunchKernelGGL(cvt_bf16_kernel,    dim3(1024),      dim3(256), 0, stream,
                     x, (uint32_t*)xbf, (long)(N_TOK * DIM / 2));
  if (prew) {
    hipLaunchKernelGGL(cvt_bf16_kernel,  dim3(8192),      dim3(256), 0, stream,
                       W1, (uint32_t*)W1b, (long)(welems / 2));
    hipLaunchKernelGGL(cvt_bf16_kernel,  dim3(8192),      dim3(256), 0, stream,
                       W2, (uint32_t*)W2b, (long)(welems / 2));
    hipLaunchKernelGGL((ffn_kernel<true>),  dim3(NEXP, 128), dim3(256), 0, stream,
                       xbf, W1, W2, W1b, W2b, ecount, tlist, wlist, rlist, partial);
  } else {
    hipLaunchKernelGGL((ffn_kernel<false>), dim3(NEXP, 128), dim3(256), 0, stream,
                       xbf, W1, W2, W1b, W2b, ecount, tlist, wlist, rlist, partial);
  }
  hipLaunchKernelGGL(combine_kernel,     dim3(2048),      dim3(256), 0, stream, partial, out);
}
